// MeanShifter_63883343561109
// MI455X (gfx1250) — matrix operand order repaired and checked by reading, not compiled, not run on hardware
//
#include <hip/hip_runtime.h>

#define D_ENC 768
#define D_MS  64
#define BATCH 8
#define NTOK  2304
#define N_LAYERS 3
#define EPS 1e-5f

typedef __bf16 bf16_t;
typedef __bf16 v16bf __attribute__((ext_vector_type(16)));
typedef __bf16 v8bf  __attribute__((ext_vector_type(8)));
typedef float  v8f   __attribute__((ext_vector_type(8)));

union Frag16 { v16bf v; v8bf h[2]; bf16_t e[16]; };

// ---------------------------------------------------------------------------
// Wq [D_ENC x D_MS] f32 -> WqT [D_MS x D_ENC] bf16
__global__ void k_wqt(const float* __restrict__ Wq, bf16_t* __restrict__ WqT) {
    int idx = blockIdx.x * 256 + threadIdx.x;     // < 768*64
    int m = idx & (D_MS - 1);
    int d = idx >> 6;
    WqT[(size_t)m * D_ENC + d] = (bf16_t)Wq[(size_t)d * D_MS + m];
}

// ---------------------------------------------------------------------------
// x [B,N,D] f32 -> xT [B,D,N] bf16 (LDS-tiled 32x32 transpose)
__global__ void k_transpose_cvt(const float* __restrict__ x, bf16_t* __restrict__ xT) {
    __shared__ float tile[32][33];
    int b  = blockIdx.z;
    int n0 = blockIdx.x * 32;
    int d0 = blockIdx.y * 32;
    int tx = threadIdx.x;   // 0..31
    int ty = threadIdx.y;   // 0..7
    const float* xb = x + (size_t)b * NTOK * D_ENC;
    for (int r = 0; r < 32; r += 8)
        tile[ty + r][tx] = xb[(size_t)(n0 + ty + r) * D_ENC + d0 + tx];
    __syncthreads();
    bf16_t* xTb = xT + (size_t)b * D_ENC * NTOK;
    for (int r = 0; r < 32; r += 8)
        xTb[(size_t)(d0 + ty + r) * NTOK + n0 + tx] = (bf16_t)tile[tx][ty + r];
}

// ---------------------------------------------------------------------------
__device__ inline void cvt8(Frag16& a, int base, const float* __restrict__ p) {
    float4 f0 = *(const float4*)(p);
    float4 f1 = *(const float4*)(p + 4);
    a.e[base + 0] = (bf16_t)f0.x; a.e[base + 1] = (bf16_t)f0.y;
    a.e[base + 2] = (bf16_t)f0.z; a.e[base + 3] = (bf16_t)f0.w;
    a.e[base + 4] = (bf16_t)f1.x; a.e[base + 5] = (bf16_t)f1.y;
    a.e[base + 6] = (bf16_t)f1.z; a.e[base + 7] = (bf16_t)f1.w;
}

// q = x @ Wq + bq  -> bf16 [B,N,64].  1 WG per 16 rows, 4 waves = 4 col tiles.
__global__ void __launch_bounds__(128) k_qproj(const float* __restrict__ x,
                                               const bf16_t* __restrict__ WqT,
                                               const float* __restrict__ bq,
                                               bf16_t* __restrict__ q) {
    int wg   = blockIdx.x;
    int b    = wg / (NTOK / 16);
    int m0   = (wg % (NTOK / 16)) * 16;
    int wave = threadIdx.x >> 5;       // 0..3 -> output col tile
    int lane = threadIdx.x & 31;
    int half = lane >> 4;
    int l15  = lane & 15;

    const float*  xrow = x + (size_t)b * NTOK * D_ENC + (size_t)(m0 + l15) * D_ENC;
    const bf16_t* wrow = WqT + (size_t)(wave * 16 + l15) * D_ENC;   // B col = l15

    v8f acc = {};
    for (int k = 0; k < D_ENC; k += 32) {
        Frag16 a, bf;
        cvt8(a, 0, xrow + k + half * 16);           // K = half*16 + 0..7
        cvt8(a, 8, xrow + k + half * 16 + 8);       // K = half*16 + 8..15
        bf.v = *(const v16bf*)(wrow + k + half * 16);
        acc = __builtin_amdgcn_wmma_f32_16x16x32_bf16(false, a.v, false, bf.v,
                                                      (short)0, acc, false, false);
    }
    float bias = bq[wave * 16 + l15];
    bf16_t* qb = q + (size_t)b * NTOK * D_MS;
#pragma unroll
    for (int i = 0; i < 8; ++i) {
        int row = m0 + i + 8 * half;
        qb[(size_t)row * D_MS + wave * 16 + l15] = (bf16_t)(acc[i] + bias);
    }
}

// ---------------------------------------------------------------------------
// Fused flash attention (q q^T softmax, P @ x) + LayerNorm.
// 1 WG per 16 query rows. 8 waves; each wave owns a 16x96 slice of the output.
#define KB 128
#define NW 8
#define DT 6

__global__ void __launch_bounds__(256) k_flash(const bf16_t* __restrict__ q,
                                               const bf16_t* __restrict__ xT,
                                               const float* __restrict__ gamma,
                                               const float* __restrict__ beta,
                                               float* __restrict__ out) {
    __shared__ __align__(16) bf16_t Pbuf[16][KB];
    __shared__ float ls_wmax[NW][16];
    __shared__ float ls_wsum[NW][16];
    __shared__ float ls_m[16], ls_l[16], ls_f[16], ls_inv[16];
    __shared__ float ls_lnsum[NW][16], ls_lnsq[NW][16];
    __shared__ float ls_mu[16], ls_rstd[16];

    int wg   = blockIdx.x;
    int b    = wg / (NTOK / 16);
    int m0   = (wg % (NTOK / 16)) * 16;
    int wave = threadIdx.x >> 5;
    int lane = threadIdx.x & 31;
    int half = lane >> 4;
    int l15  = lane & 15;

    const bf16_t* qb  = q  + (size_t)b * NTOK * D_MS;
    const bf16_t* xTb = xT + (size_t)b * D_ENC * NTOK;

    if (threadIdx.x < 16) { ls_m[threadIdx.x] = -__builtin_inff(); ls_l[threadIdx.x] = 0.f; }
    if (threadIdx.x < NW * 16) ((float*)ls_wsum)[threadIdx.x] = 0.f;

    // Persistent A fragments: q rows m0..m0+15, K = 0..63
    Frag16 aq0, aq1;
    const bf16_t* qrow = qb + (size_t)(m0 + l15) * D_MS;
    aq0.h[0] = *(const v8bf*)(qrow + half * 16);
    aq0.h[1] = *(const v8bf*)(qrow + half * 16 + 8);
    aq1.h[0] = *(const v8bf*)(qrow + 32 + half * 16);
    aq1.h[1] = *(const v8bf*)(qrow + 32 + half * 16 + 8);

    v8f zero = {};
    v8f acc[DT];
#pragma unroll
    for (int t = 0; t < DT; ++t) acc[t] = zero;

    const float scale = 0.125f;        // 1/sqrt(64)
    int d0w = wave * (DT * 16);        // 96 cols per wave

    for (int kb = 0; kb < NTOK; kb += KB) {
        // ---- S tile: keys kb + wave*16 .. +15
        int key0 = kb + wave * 16;
        const bf16_t* krow = qb + (size_t)(key0 + l15) * D_MS;   // B col = l15
        Frag16 b0, b1;
        b0.v = *(const v16bf*)(krow + half * 16);
        b1.v = *(const v16bf*)(krow + 32 + half * 16);
        v8f s = zero;
        s = __builtin_amdgcn_wmma_f32_16x16x32_bf16(false, aq0.v, false, b0.v, (short)0, s, false, false);
        s = __builtin_amdgcn_wmma_f32_16x16x32_bf16(false, aq1.v, false, b1.v, (short)0, s, false, false);

        float sv[8], tmax[8];
#pragma unroll
        for (int i = 0; i < 8; ++i) { sv[i] = s[i] * scale; tmax[i] = sv[i]; }
#pragma unroll
        for (int mk = 1; mk < 16; mk <<= 1)
#pragma unroll
            for (int i = 0; i < 8; ++i) tmax[i] = fmaxf(tmax[i], __shfl_xor(tmax[i], mk, 32));
        if (l15 == 0)
#pragma unroll
            for (int i = 0; i < 8; ++i) ls_wmax[wave][i + 8 * half] = tmax[i];
        __syncthreads();                                          // (B)
        if (threadIdx.x < 16) {                                   // online-softmax stats
            int r = threadIdx.x;
            float bm = ls_wmax[0][r];
#pragma unroll
            for (int w = 1; w < NW; ++w) bm = fmaxf(bm, ls_wmax[w][r]);
            float lsum = 0.f;
#pragma unroll
            for (int w = 0; w < NW; ++w) lsum += ls_wsum[w][r];   // prev block sums
            float m_old = ls_m[r];
            float m_new = fmaxf(m_old, bm);
            float f = __expf(m_old - m_new);
            ls_f[r] = f;
            ls_l[r] = (ls_l[r] + lsum) * f;
            ls_m[r] = m_new;
        }
        __syncthreads();                                          // (C)
        float fac[8], mrow[8];
#pragma unroll
        for (int i = 0; i < 8; ++i) { fac[i] = ls_f[i + 8 * half]; mrow[i] = ls_m[i + 8 * half]; }
#pragma unroll
        for (int t = 0; t < DT; ++t)
#pragma unroll
            for (int i = 0; i < 8; ++i) acc[t][i] *= fac[i];
        float psum[8];
#pragma unroll
        for (int i = 0; i < 8; ++i) {
            float p = __expf(sv[i] - mrow[i]);
            psum[i] = p;
            Pbuf[i + 8 * half][wave * 16 + l15] = (bf16_t)p;
        }
#pragma unroll
        for (int mk = 1; mk < 16; mk <<= 1)
#pragma unroll
            for (int i = 0; i < 8; ++i) psum[i] += __shfl_xor(psum[i], mk, 32);
        if (l15 == 0)
#pragma unroll
            for (int i = 0; i < 8; ++i) ls_wsum[wave][i + 8 * half] = psum[i];
        __syncthreads();                                          // (D)

        // ---- acc += P[16xKB] @ xT block
#pragma unroll
        for (int kk = 0; kk < KB; kk += 32) {
            Frag16 ap;
            ap.h[0] = *(const v8bf*)(&Pbuf[l15][kk + half * 16]);
            ap.h[1] = *(const v8bf*)(&Pbuf[l15][kk + half * 16 + 8]);
#pragma unroll
            for (int t = 0; t < DT; ++t) {
                int d = d0w + t * 16 + l15;
                Frag16 bx;
                bx.v = *(const v16bf*)(xTb + (size_t)d * NTOK + kb + kk + half * 16);
                acc[t] = __builtin_amdgcn_wmma_f32_16x16x32_bf16(false, ap.v, false, bx.v,
                                                                 (short)0, acc[t], false, false);
            }
        }
    }

    // ---- epilogue: 1/l normalize, then LayerNorm over D_ENC
    if (threadIdx.x < 16) {
        int r = threadIdx.x;
        float lt = ls_l[r];
#pragma unroll
        for (int w = 0; w < NW; ++w) lt += ls_wsum[w][r];
        ls_inv[r] = 1.0f / lt;
    }
    __syncthreads();
    float inv[8];
#pragma unroll
    for (int i = 0; i < 8; ++i) inv[i] = ls_inv[i + 8 * half];
    float rsum[8], rsq[8];
#pragma unroll
    for (int i = 0; i < 8; ++i) { rsum[i] = 0.f; rsq[i] = 0.f; }
#pragma unroll
    for (int t = 0; t < DT; ++t)
#pragma unroll
        for (int i = 0; i < 8; ++i) {
            float v = acc[t][i] * inv[i];
            acc[t][i] = v;
            rsum[i] += v; rsq[i] += v * v;
        }
#pragma unroll
    for (int mk = 1; mk < 16; mk <<= 1)
#pragma unroll
        for (int i = 0; i < 8; ++i) {
            rsum[i] += __shfl_xor(rsum[i], mk, 32);
            rsq[i]  += __shfl_xor(rsq[i],  mk, 32);
        }
    if (l15 == 0)
#pragma unroll
        for (int i = 0; i < 8; ++i) { ls_lnsum[wave][i + 8 * half] = rsum[i]; ls_lnsq[wave][i + 8 * half] = rsq[i]; }
    __syncthreads();
    if (threadIdx.x < 16) {
        int r = threadIdx.x;
        float s = 0.f, s2 = 0.f;
#pragma unroll
        for (int w = 0; w < NW; ++w) { s += ls_lnsum[w][r]; s2 += ls_lnsq[w][r]; }
        float mu  = s * (1.0f / D_ENC);
        float var = s2 * (1.0f / D_ENC) - mu * mu;
        ls_mu[r] = mu;
        ls_rstd[r] = rsqrtf(var + EPS);
    }
    __syncthreads();
    float* ob = out + (size_t)b * NTOK * D_ENC;
#pragma unroll
    for (int t = 0; t < DT; ++t) {
        int d = d0w + t * 16 + l15;
        float g = gamma[d], be = beta[d];
#pragma unroll
        for (int i = 0; i < 8; ++i) {
            int row = m0 + i + 8 * half;
            ob[(size_t)row * D_ENC + d] =
                (acc[t][i] - ls_mu[i + 8 * half]) * ls_rstd[i + 8 * half] * g + be;
        }
    }
}

// ---------------------------------------------------------------------------
extern "C" void kernel_launch(void* const* d_in, const int* in_sizes, int n_in,
                              void* d_out, int out_size, void* d_ws, size_t ws_size,
                              hipStream_t stream) {
    (void)in_sizes; (void)n_in; (void)out_size; (void)ws_size;
    const float* x_in  = (const float*)d_in[0];
    const float* Wq    = (const float*)d_in[1];
    const float* bqv   = (const float*)d_in[2];
    const float* gamma = (const float*)d_in[3];
    const float* beta  = (const float*)d_in[4];

    char* ws = (char*)d_ws;
    size_t off = 0;
    bf16_t* xT  = (bf16_t*)(ws + off); off += (size_t)BATCH * D_ENC * NTOK * sizeof(bf16_t);
    bf16_t* qb  = (bf16_t*)(ws + off); off += (size_t)BATCH * NTOK * D_MS * sizeof(bf16_t);
    bf16_t* WqT = (bf16_t*)(ws + off); off += (size_t)D_MS * D_ENC * sizeof(bf16_t);
    off = (off + 255) & ~(size_t)255;
    float* xmid = (float*)(ws + off);

    k_wqt<<<(D_ENC * D_MS) / 256, 256, 0, stream>>>(Wq, WqT);

    const float* cur = x_in;
    float* outs[3] = { (float*)d_out, xmid, (float*)d_out };
    for (int L = 0; L < N_LAYERS; ++L) {
        dim3 tg(NTOK / 32, D_ENC / 32, BATCH);
        k_transpose_cvt<<<tg, dim3(32, 8), 0, stream>>>(cur, xT);
        k_qproj<<<BATCH * (NTOK / 16), 128, 0, stream>>>(cur, WqT, bqv, qb);
        k_flash<<<BATCH * (NTOK / 16), 256, 0, stream>>>(qb, xT, gamma, beta, outs[L]);
        cur = outs[L];
    }
}
